// AugmentedGraphNeuralODEFunc_91293824844478
// MI455X (gfx1250) — compile-verified, matrix operand order checked by reading, not run on hardware
//
#include <hip/hip_runtime.h>

typedef __attribute__((ext_vector_type(16))) _Float16 v16h;
typedef __attribute__((ext_vector_type(8)))  float    v8f;
typedef __attribute__((ext_vector_type(4)))  float    f32x4;

#define ORIG 16
#define AUG  32
#define HID  128
#define NB   4
#define NN   512
#define LDW  136   // padded LDS row stride in f16 elements (272B) -> conflict-free b128 frag loads

// ---------------------------------------------------------------------------
// Prep: transpose edge-MLP weights to N-major f16, and fold x_i / x_j first-
// layer contributions into per-node vectors P[i] = x_i @ W0[0:32],
// Q[j] = x_j @ W0[32:64] + b0.
// ---------------------------------------------------------------------------
__global__ void prep_kernel(const float* __restrict__ z,
                            const float* __restrict__ eW0,
                            const float* __restrict__ eb0,
                            const float* __restrict__ eW1,
                            const float* __restrict__ eW2,
                            float* __restrict__ P, float* __restrict__ Q,
                            _Float16* __restrict__ W1t, _Float16* __restrict__ W2t)
{
    const int tid = blockIdx.x * blockDim.x + threadIdx.x;
    const int T   = gridDim.x * blockDim.x;

    // W1t[n][k] = eW1[k][n]  (128x128)
    for (int idx = tid; idx < HID * HID; idx += T) {
        int n = idx / HID, k = idx - n * HID;
        W1t[idx] = (_Float16)eW1[k * HID + n];
    }
    // W2t[n][k] = eW2[k][n]  (16x128)
    for (int idx = tid; idx < ORIG * HID; idx += T) {
        int n = idx / HID, k = idx - n * HID;
        W2t[idx] = (_Float16)eW2[k * ORIG + n];
    }
    // P, Q per node (2048 x 128)
    for (int idx = tid; idx < NB * NN * HID; idx += T) {
        int r = idx / HID, c = idx - r * HID;
        const float* zr = z + r * AUG;
        float p = 0.f, q = eb0[c];
        #pragma unroll 8
        for (int k = 0; k < AUG; ++k) {
            p += zr[k] * eW0[k * HID + c];
            q += zr[k] * eW0[(AUG + k) * HID + c];
        }
        P[idx] = p;
        Q[idx] = q;
    }
}

// load one 16x16x32 f16 fragment row-slice: p points at rowbase + kk*32 + half*8
__device__ __forceinline__ v16h load_frag(const _Float16* p)
{
    union { v16h v; f32x4 f[2]; } u;
    const f32x4* q = (const f32x4*)p;
    u.f[0] = q[0];   // K = base .. base+7
    u.f[1] = q[2];   // K = base+16 .. base+23
    return u.v;
}

// ---------------------------------------------------------------------------
// Edge kernel: one workgroup = 16 i x 16 j tile (256 edges), 16 waves.
// ---------------------------------------------------------------------------
__launch_bounds__(512, 1)
__global__ void edge_kernel(const float* __restrict__ z,
                            const float* __restrict__ eW0,
                            const float* __restrict__ eb1,
                            const float* __restrict__ eb2,
                            const float* __restrict__ P,
                            const float* __restrict__ Q,
                            const _Float16* __restrict__ W1t,
                            const _Float16* __restrict__ W2t,
                            float* __restrict__ aggp)
{
    __shared__ __align__(16) _Float16 sH[256 * LDW];   // 69632 B  (h0 -> h1 in place)
    __shared__ __align__(16) _Float16 sW1[HID * LDW];  // 34816 B
    __shared__ __align__(16) _Float16 sW2[ORIG * LDW]; //  4352 B
    __shared__ float sW0g[8 * HID];                    //  4096 B (geo rows 64..71 of eW0)
    __shared__ float sP[16 * HID];                     //  8192 B
    __shared__ float sQ[16 * HID];                     //  8192 B
    __shared__ float sb1[HID];
    __shared__ float sb2[ORIG];
    __shared__ float sZi[16][8];
    __shared__ float sZj[16][8];

    const int tid = threadIdx.x;
    const int b   = blockIdx.z;
    const int it  = blockIdx.y;
    const int jt  = blockIdx.x;
    const int i0  = it * 16, j0 = jt * 16;

    // ---- cooperative loads ----
    for (int x = tid; x < HID * HID; x += 512) {
        int n = x / HID, k = x - n * HID;
        sW1[n * LDW + k] = W1t[x];
    }
    for (int x = tid; x < ORIG * HID; x += 512) {
        int n = x / HID, k = x - n * HID;
        sW2[n * LDW + k] = W2t[x];
    }
    for (int x = tid; x < 8 * HID; x += 512) sW0g[x] = eW0[64 * HID + x];
    for (int x = tid; x < 16 * HID; x += 512) {
        sP[x] = P[(b * NN + i0) * HID + x];
        sQ[x] = Q[(b * NN + j0) * HID + x];
    }
    if (tid < HID)  sb1[tid] = eb1[tid];
    if (tid < ORIG) sb2[tid] = eb2[tid];
    if (tid < 32) {
        int r = tid & 15;
        const float* zr = z + (size_t)(b * NN + ((tid >= 16) ? j0 : i0) + r) * AUG;
        float* dz = (tid >= 16) ? sZj[r] : sZi[r];
        dz[0] = zr[0];  dz[1] = zr[1];  dz[2] = zr[2];     // pos   (evolving[:3])
        dz[3] = zr[16]; dz[4] = zr[17]; dz[5] = zr[18];    // pos_s (static[:3])
    }
    __syncthreads();

    // ---- layer 0: h0 = relu(P[i] + Q[j] + geo @ W0g)  (b0 folded into Q) ----
    {
        const int e = tid >> 1, part = tid & 1;
        const int li = e >> 4, lj = e & 15;
        float dx = sZi[li][0] - sZj[lj][0];
        float dy = sZi[li][1] - sZj[lj][1];
        float dzv = sZi[li][2] - sZj[lj][2];
        float sq = dx * dx + dy * dy + dzv * dzv;
        float dist = (sq > 0.f) ? sqrtf(sq) : 0.f;
        float ex = sZi[li][3] - sZj[lj][3];
        float ey = sZi[li][4] - sZj[lj][4];
        float ez = sZi[li][5] - sZj[lj][5];
        float sqs = ex * ex + ey * ey + ez * ez;
        float dists = (sqs > 0.f) ? sqrtf(sqs) : 0.f;
        float g[8] = { dx, dy, dzv, dist, ex, ey, ez, dists };

        _Float16* hrow = sH + e * LDW;
        const float* pr = sP + li * HID;
        const float* qr = sQ + lj * HID;
        const int c0 = part * 64;
        #pragma unroll 4
        for (int c = c0; c < c0 + 64; ++c) {
            float h = pr[c] + qr[c];
            #pragma unroll
            for (int k = 0; k < 8; ++k) h += g[k] * sW0g[k * HID + c];
            hrow[c] = (_Float16)(h > 0.f ? h : 0.f);
        }
    }
    __syncthreads();

    const int wv   = tid >> 5;       // wave = M-tile (fixed i = i0+wv, 16 j's as rows)
    const int lane = tid & 31;
    const int lm   = lane & 15;
    const int half = lane >> 4;

    const _Float16* arow = sH + (wv * 16 + lm) * LDW + half * 8;

    // ---- layer 1: H1 = relu(H0 @ W1 + b1), WMMA 16x16x32, in-place rows ----
    v8f acc[8];
    #pragma unroll
    for (int n = 0; n < 8; ++n)
        #pragma unroll
        for (int r = 0; r < 8; ++r) acc[n][r] = 0.f;

    #pragma unroll
    for (int kk = 0; kk < 4; ++kk) {
        v16h a = load_frag(arow + kk * 32);
        #pragma unroll
        for (int n = 0; n < 8; ++n) {
            v16h bf = load_frag(sW1 + (n * 16 + lm) * LDW + half * 8 + kk * 32);
            acc[n] = __builtin_amdgcn_wmma_f32_16x16x32_f16(
                false, a, false, bf, (short)0, acc[n], false, false);
        }
    }
    #pragma unroll
    for (int n = 0; n < 8; ++n) {
        const float bias = sb1[n * 16 + lm];
        #pragma unroll
        for (int r = 0; r < 8; ++r) {
            float v = acc[n][r] + bias;
            v = (v > 0.f) ? v : 0.f;
            const int row = wv * 16 + r + half * 8;
            sH[row * LDW + n * 16 + lm] = (_Float16)v;   // same-wave rows: DS in-order
        }
    }

    // ---- layer 2: msg = H1 @ W2 (+b2), then sum rows (= sum over j) ----
    v8f c2;
    #pragma unroll
    for (int r = 0; r < 8; ++r) c2[r] = 0.f;
    #pragma unroll
    for (int kk = 0; kk < 4; ++kk) {
        v16h a  = load_frag(arow + kk * 32);
        v16h bf = load_frag(sW2 + lm * LDW + half * 8 + kk * 32);
        c2 = __builtin_amdgcn_wmma_f32_16x16x32_f16(
            false, a, false, bf, (short)0, c2, false, false);
    }
    float s = 0.f;
    #pragma unroll
    for (int r = 0; r < 8; ++r) s += c2[r];   // sum over 8 of the 16 M-rows
    s += __shfl_xor(s, 16, 32);               // combine both halves -> full row sum
    if (lane < 16) {
        // each of 16 edges contributed +b2 -> +16*b2 per tile
        const float val = s + 16.f * sb2[lane];
        const int node = b * NN + i0 + wv;
        aggp[((size_t)node * 32 + jt) * ORIG + lane] = val;   // deterministic partials
    }
}

// ---------------------------------------------------------------------------
// Node update: deterministic reduction of partials + small fp32 MLP.
// ---------------------------------------------------------------------------
__global__ void node_kernel(const float* __restrict__ z,
                            const float* __restrict__ aggp,
                            const float* __restrict__ uW0, const float* __restrict__ ub0,
                            const float* __restrict__ uW1, const float* __restrict__ ub1,
                            const float* __restrict__ uW2, const float* __restrict__ ub2,
                            float* __restrict__ out)
{
    __shared__ float xin[2 * ORIG];
    __shared__ float h0[HID];
    __shared__ float h1[HID];

    const int node = blockIdx.x;     // 0..2047
    const int tid  = threadIdx.x;    // 128

    if (tid < 2 * ORIG) {
        if (tid < ORIG) {
            xin[tid] = z[(size_t)node * AUG + tid];       // evolving
        } else {
            const float* pp = aggp + (size_t)node * 32 * ORIG + (tid - ORIG);
            float sv = 0.f;
            for (int j = 0; j < 32; ++j) sv += pp[j * ORIG];  // fixed order
            xin[tid] = sv;
        }
    }
    __syncthreads();
    {
        float a = ub0[tid];
        #pragma unroll 8
        for (int k = 0; k < 2 * ORIG; ++k) a += xin[k] * uW0[k * HID + tid];
        h0[tid] = (a > 0.f) ? a : 0.f;
    }
    __syncthreads();
    {
        float a = ub1[tid];
        #pragma unroll 8
        for (int k = 0; k < HID; ++k) a += h0[k] * uW1[k * HID + tid];
        h1[tid] = (a > 0.f) ? a : 0.f;
    }
    __syncthreads();
    if (tid < ORIG) {
        float a = ub2[tid];
        #pragma unroll 8
        for (int k = 0; k < HID; ++k) a += h1[k] * uW2[k * ORIG + tid];
        out[(size_t)node * AUG + tid]        = a;    // d_evolving
        out[(size_t)node * AUG + ORIG + tid] = 0.f;  // d_static
    }
}

// ---------------------------------------------------------------------------
extern "C" void kernel_launch(void* const* d_in, const int* in_sizes, int n_in,
                              void* d_out, int out_size, void* d_ws, size_t ws_size,
                              hipStream_t stream)
{
    (void)in_sizes; (void)n_in; (void)out_size; (void)ws_size;
    const float* z   = (const float*)d_in[1];
    const float* eW0 = (const float*)d_in[2];
    const float* eb0 = (const float*)d_in[3];
    const float* eW1 = (const float*)d_in[4];
    const float* eb1 = (const float*)d_in[5];
    const float* eW2 = (const float*)d_in[6];
    const float* eb2 = (const float*)d_in[7];
    const float* uW0 = (const float*)d_in[8];
    const float* ub0 = (const float*)d_in[9];
    const float* uW1 = (const float*)d_in[10];
    const float* ub1 = (const float*)d_in[11];
    const float* uW2 = (const float*)d_in[12];
    const float* ub2 = (const float*)d_in[13];
    float* out = (float*)d_out;

    char* ws = (char*)d_ws;
    float*    P    = (float*)ws;  ws += (size_t)NB * NN * HID * 4;      // 1 MB
    float*    Qv   = (float*)ws;  ws += (size_t)NB * NN * HID * 4;      // 1 MB
    float*    aggp = (float*)ws;  ws += (size_t)NB * NN * 32 * ORIG * 4;// 4 MB partials
    _Float16* W1t  = (_Float16*)ws; ws += (size_t)HID * HID * 2;        // 32 KB
    _Float16* W2t  = (_Float16*)ws; ws += (size_t)ORIG * HID * 2;       // 4 KB

    prep_kernel<<<512, 256, 0, stream>>>(z, eW0, eb0, eW1, eW2, P, Qv, W1t, W2t);

    dim3 grid(NN / 16, NN / 16, NB);   // (jt, it, b) = 32 x 32 x 4
    edge_kernel<<<grid, 512, 0, stream>>>(z, eW0, eb1, eb2, P, Qv, W1t, W2t, aggp);

    node_kernel<<<NB * NN, HID, 0, stream>>>(z, aggp, uW0, ub0, uW1, ub1, uW2, ub2, out);
}
